// Predictor_15118284882401
// MI455X (gfx1250) — matrix operand order repaired and checked by reading, not compiled, not run on hardware
//
#include <hip/hip_runtime.h>

// ---------------------------------------------------------------------------
// PredRNN forward for MI455X (gfx1250): implicit-GEMM convolutions on
// v_wmma_f32_16x16x32_bf16 (wave32), fp32 accumulate, fused epilogues,
// async global->LDS staging of the weight tile (ASYNCcnt path),
// 64-wide K staging rounds (4 WMMAs per wave between barriers).
// ---------------------------------------------------------------------------

typedef __attribute__((ext_vector_type(8)))  float  v8f;
typedef __attribute__((ext_vector_type(16))) __bf16 v16bf;

#if defined(__gfx1250__) && __has_builtin(__builtin_amdgcn_global_load_async_to_lds_b128) && \
    __has_builtin(__builtin_amdgcn_s_wait_asynccnt)
#define USE_ASYNC_LDS 1
typedef __attribute__((__vector_size__(16))) int v4i_t;
typedef __attribute__((address_space(1))) v4i_t as1_v4i;   // global v4i
typedef __attribute__((address_space(3))) v4i_t as3_v4i;   // LDS v4i
#else
#define USE_ASYNC_LDS 0
#endif

static __device__ __forceinline__ unsigned short f32_to_bf16_rne(float f) {
  unsigned u = __float_as_uint(f);
  unsigned r = u + 0x7FFFu + ((u >> 16) & 1u);
  return (unsigned short)(r >> 16);
}
static __device__ __forceinline__ float sigm(float x) { return 1.0f / (1.0f + expf(-x)); }

// ---------------------------------------------------------------------------
// Weight packing: fp32 OIHW -> bf16 [Cout][KP] row-major (k = ci*kh*kw+r*kw+s,
// identical to OIHW flat order), zero-padded to KP (multiple of 64).
// ---------------------------------------------------------------------------
__global__ void pack_w(const float* __restrict__ w, unsigned short* __restrict__ wp,
                       int Ktot, int KP, int Cout) {
  int i = blockIdx.x * blockDim.x + threadIdx.x;
  int total = Cout * KP;
  if (i >= total) return;
  int co = i / KP;
  int k  = i - co * KP;
  float v = (k < Ktot) ? w[(size_t)co * Ktot + k] : 0.0f;
  wp[i] = f32_to_bf16_rne(v);
}

// y = relu(acc*sc + sh), sc = g*rsqrt(var+eps), sh = (bconv-mu)*sc + beta
__global__ void bn_prep(const float* __restrict__ bconv, const float* __restrict__ g,
                        const float* __restrict__ beta, const float* __restrict__ mu,
                        const float* __restrict__ var, float* __restrict__ sc,
                        float* __restrict__ sh, int C) {
  int i = blockIdx.x * blockDim.x + threadIdx.x;
  if (i >= C) return;
  float s = g[i] * rsqrtf(var[i] + 1e-5f);
  sc[i] = s;
  sh[i] = (bconv[i] - mu[i]) * s + beta[i];
}

__global__ void fillz(float* __restrict__ p, size_t n) {
  size_t i = (size_t)blockIdx.x * blockDim.x + threadIdx.x;
  if (i < n) p[i] = 0.0f;
}

// ---------------------------------------------------------------------------
// Implicit-GEMM conv. Block: 256 thr = 8 waves. Tile: 64 Cout x 64 positions.
// Wave (wid): co group = wid>>1 (16 rows), pos group = wid&1 (32 cols).
// Each staging round covers K=64: 2 K-sub-chunks x 2 pos tiles = 4 WMMAs/wave.
// LDS row strides padded to kill bank conflicts; A rows 16B-aligned for b128.
// ---------------------------------------------------------------------------
struct ConvParams {
  const float* in;
  const unsigned short* wp;   // bf16 [Cout][KP]
  const float* p0;            // bias (mode 0) or BN scale (mode 1)
  const float* p1;            // BN shift (mode 1)
  float* out;
  int Cin, Hin, Win, Cout, Hout, Wout;
  int stride, pad;
  int Ktot, KP;
  int inBatchStride;          // elements per batch in input tensor
  int inChanOff;              // extra element offset (frame slicing)
};

#define AS_STRIDE 72   // ushorts per A row (144B, 16B aligned) -> 36 dwords
#define BS_STRIDE 68   // ushorts per B row (136B)              -> 34 dwords

template <int KH, int KW, int MODE>
__global__ __launch_bounds__(256) void conv_wmma(ConvParams cp) {
  __shared__ __attribute__((aligned(16))) unsigned short As[64 * AS_STRIDE];
  __shared__ __attribute__((aligned(16))) unsigned short Bs[64 * BS_STRIDE];

  const int tid  = threadIdx.x;
  const int lane = tid & 31;
  const int wid  = tid >> 5;
  const int l15  = lane & 15;
  const int hi   = lane >> 4;
  const int coW  = (wid >> 1) * 16;        // wave's co offset within 64-tile
  const int posW = (wid & 1) * 32;         // wave's pos offset within 64-tile

  const int posTile = blockIdx.x * 64;
  const int coTile  = blockIdx.y * 64;
  const int HWin    = cp.Hin * cp.Win;
  const int HWo     = cp.Hout * cp.Wout;

  // ---- loop-invariant staging coordinates (hoisted out of the K-loop) ----
  // A stage: two 16B chunks per thread (row is 64 bf16 = 128B, 4 thr/row).
  const int rowA = tid >> 2;               // 0..63 co row
  const int kk0A = (tid & 3) * 8;          // 0,8,16,24 (+32 for second chunk)
  // B stage: one position per thread, 16 k-slots per round.
  const int pB  = tid & 63;                // position within tile
  const int kkB = tid >> 6;                // 0..3 (step 4)
  const int pos = posTile + pB;
  const int bB  = pos / HWo;
  const int prB = pos - bB * HWo;
  const int pyB = prB / cp.Wout;
  const int pxB = prB - pyB * cp.Wout;
  const int iy0 = pyB * cp.stride - cp.pad;
  const int ix0 = pxB * cp.stride - cp.pad;
  const float* __restrict__ inB = cp.in + (size_t)bB * cp.inBatchStride + cp.inChanOff;

  v8f acc0 = {};
  v8f acc1 = {};

  for (int kc = 0; kc < cp.KP; kc += 64) {
    // ---- stage A: 64x64 bf16 weights (pure copy -> async LDS engine) ----
    {
      const unsigned short* gA = cp.wp + (size_t)(coTile + rowA) * cp.KP + kc + kk0A;
      unsigned short* lA = &As[rowA * AS_STRIDE + kk0A];
#if USE_ASYNC_LDS
      __builtin_amdgcn_global_load_async_to_lds_b128(
          (as1_v4i*)(size_t)gA, (as3_v4i*)(unsigned)(size_t)lA, 0, 0);
      __builtin_amdgcn_global_load_async_to_lds_b128(
          (as1_v4i*)(size_t)(gA + 32), (as3_v4i*)(unsigned)(size_t)(lA + 32), 0, 0);
#else
      *(uint4*)lA = *(const uint4*)gA;
      *(uint4*)(lA + 32) = *(const uint4*)(gA + 32);
#endif
      if (kc + 64 < cp.KP)  // CDNA5 prefetch path (global_prefetch_b8)
        __builtin_prefetch(gA + 64, 0, 0);
    }
    // ---- stage B: im2col 64 pos x 64 k, fp32 -> bf16 (overlaps async A) ----
#pragma unroll
    for (int j = 0; j < 16; ++j) {
      const int kk = kkB + 4 * j;
      const int k  = kc + kk;
      float val = 0.0f;
      if (k < cp.Ktot) {
        const int ci  = k / (KH * KW);          // compile-time divisor
        const int rem = k - ci * (KH * KW);
        const int r   = rem / KW;
        const int s   = rem - r * KW;
        const int iy  = iy0 + r;
        const int ix  = ix0 + s;
        if ((unsigned)iy < (unsigned)cp.Hin && (unsigned)ix < (unsigned)cp.Win)
          val = inB[(size_t)ci * HWin + (size_t)iy * cp.Win + ix];
      }
      Bs[pB * BS_STRIDE + kk] = f32_to_bf16_rne(val);
    }
#if USE_ASYNC_LDS
    __builtin_amdgcn_s_wait_asynccnt(0);
#endif
    __syncthreads();

    // ---- 2 K-sub-chunks x 2 pos tiles = 4 WMMAs per wave ----
    const unsigned* As32 = (const unsigned*)As;   // row stride 36 dwords
    const unsigned* Bs32 = (const unsigned*)Bs;   // row stride 34 dwords
    const int arow  = (coW + l15) * (AS_STRIDE / 2);
    const int brow0 = (posW + l15) * (BS_STRIDE / 2);
    const int brow1 = (posW + 16 + l15) * (BS_STRIDE / 2);
#pragma unroll
    for (int sub = 0; sub < 2; ++sub) {
      union Frag { v16bf v; unsigned u[8]; };
      Frag a, b0, b1;
      const int soff = 16 * sub;                  // +32 k = +16 dwords
#pragma unroll
      for (int vv = 0; vv < 8; ++vv) {
        // A (16x32 bf16): k = 16*hi + 2*vv  -> dword index
        const int adw = 8 * hi + vv;
        a.u[vv]  = As32[arow + soff + adw];
        // B (32x16 bf16): k = 16*hi + 2*vv  -> dword index
        const int bdw = 8 * hi + vv;
        b0.u[vv] = Bs32[brow0 + soff + bdw];
        b1.u[vv] = Bs32[brow1 + soff + bdw];
      }
      acc0 = __builtin_amdgcn_wmma_f32_16x16x32_bf16(false, a.v, false, b0.v,
                                                     (short)0, acc0, false, false);
      acc1 = __builtin_amdgcn_wmma_f32_16x16x32_bf16(false, a.v, false, b1.v,
                                                     (short)0, acc1, false, false);
    }
    __syncthreads();
  }

  // ---- epilogue: C/D layout -> lane l: n = l&15, m = r + 8*(l>>4) ----
#pragma unroll
  for (int t = 0; t < 2; ++t) {
    v8f accv = t ? acc1 : acc0;
    const int n  = posTile + posW + t * 16 + l15;
    const int bb = n / HWo;
    const int pr = n - bb * HWo;
#pragma unroll
    for (int r = 0; r < 8; ++r) {
      const int co = coTile + coW + r + 8 * hi;
      float v = accv[r];
      if (MODE == 1) v = fmaxf(v * cp.p0[co] + cp.p1[co], 0.0f);
      else           v = v + cp.p0[co];
      cp.out[((size_t)bb * cp.Cout + co) * HWo + pr] = v;
    }
  }
}

// ---------------------------------------------------------------------------
// ST-LSTM gate math (elementwise, fp32). Updates c,m in place, builds
// mem = concat(c_new, m_new) and opre = ox + oh.
// ---------------------------------------------------------------------------
__global__ void gates_kernel(const float* __restrict__ xc, const float* __restrict__ hc,
                             const float* __restrict__ mc, float* __restrict__ c_io,
                             float* __restrict__ m_io, float* __restrict__ mem,
                             float* __restrict__ opre, int Bn, int C, int SS) {
  const size_t total = (size_t)Bn * C * SS;
  const size_t i = (size_t)blockIdx.x * blockDim.x + threadIdx.x;
  if (i >= total) return;
  const size_t CSS = (size_t)C * SS;
  const size_t b   = i / CSS;
  const size_t rem = i - b * CSS;       // cch*SS + spatial
  const size_t xb = b * 7 * CSS + rem;
  const size_t hb = b * 4 * CSS + rem;
  const size_t mb = b * 3 * CSS + rem;

  const float ix  = xc[xb],            fx  = xc[xb + CSS],     gx  = xc[xb + 2 * CSS];
  const float ixp = xc[xb + 3 * CSS],  fxp = xc[xb + 4 * CSS], gxp = xc[xb + 5 * CSS];
  const float ox  = xc[xb + 6 * CSS];
  const float ih  = hc[hb],            fh  = hc[hb + CSS];
  const float gh  = hc[hb + 2 * CSS],  oh  = hc[hb + 3 * CSS];
  const float im  = mc[mb],            fm  = mc[mb + CSS],     gm  = mc[mb + 2 * CSS];

  const float ii = sigm(ix + ih);
  const float ff = sigm(fx + fh);
  const float gg = tanhf(gx + gh);
  const float cn = ff * c_io[i] + ii * gg;

  const float ip  = sigm(ixp + im);
  const float fp_ = sigm(fxp + fm);
  const float gp  = tanhf(gxp + gm);
  const float mn  = fp_ * m_io[i] + ip * gp;

  c_io[i] = cn;
  m_io[i] = mn;
  const size_t mi = b * 2 * CSS + rem;
  mem[mi]       = cn;
  mem[mi + CSS] = mn;
  opre[i] = ox + oh;
}

__global__ void hout_kernel(const float* __restrict__ opre, const float* __restrict__ oc,
                            const float* __restrict__ hl, float* __restrict__ h,
                            size_t n) {
  const size_t i = (size_t)blockIdx.x * blockDim.x + threadIdx.x;
  if (i >= n) return;
  h[i] = sigm(opre[i] + oc[i]) * tanhf(hl[i]);
}

// ---------------------------------------------------------------------------
// Host orchestration
// ---------------------------------------------------------------------------
static void launch_conv(hipStream_t s, const float* in, int inBatchStride, int inChanOff,
                        const unsigned short* wp, const float* p0, const float* p1,
                        float* out, int Bn, int Cin, int Hin, int Win, int Cout,
                        int kh, int kw, int stride, int pad, int mode) {
  ConvParams cp;
  cp.in = in; cp.wp = wp; cp.p0 = p0; cp.p1 = p1; cp.out = out;
  cp.Cin = Cin; cp.Hin = Hin; cp.Win = Win; cp.Cout = Cout;
  cp.Hout = (Hin + 2 * pad - kh) / stride + 1;
  cp.Wout = (Win + 2 * pad - kw) / stride + 1;
  cp.stride = stride; cp.pad = pad;
  cp.Ktot = Cin * kh * kw;
  cp.KP = ((cp.Ktot + 63) / 64) * 64;
  cp.inBatchStride = inBatchStride; cp.inChanOff = inChanOff;
  const int Npos = Bn * cp.Hout * cp.Wout;     // multiple of 64 for all shapes here
  dim3 grid(Npos / 64, Cout / 64);
  if (kh == 3 && mode == 1)      conv_wmma<3, 3, 1><<<grid, 256, 0, s>>>(cp);
  else if (kh == 3)              conv_wmma<3, 3, 0><<<grid, 256, 0, s>>>(cp);
  else                           conv_wmma<1, 1, 0><<<grid, 256, 0, s>>>(cp);
}

extern "C" void kernel_launch(void* const* d_in, const int* in_sizes, int n_in,
                              void* d_out, int out_size, void* d_ws, size_t ws_size,
                              hipStream_t stream) {
  (void)in_sizes; (void)n_in; (void)out_size; (void)ws_size;
  const int Bn = 16, C = 128, S = 64, Himg = 256;

  // -------- input pointers (setup_inputs insertion order) --------
  int idx = 0;
  const float* x = (const float*)d_in[idx++];
  const float *ew[4], *eb[4], *eg[4], *ebe[4], *emu[4], *evr[4];
  for (int l = 0; l < 4; ++l) {
    ew[l]  = (const float*)d_in[idx++];
    eb[l]  = (const float*)d_in[idx++];
    eg[l]  = (const float*)d_in[idx++];
    ebe[l] = (const float*)d_in[idx++];
    emu[l] = (const float*)d_in[idx++];
    evr[l] = (const float*)d_in[idx++];
  }
  const float *wx[4], *bx[4], *wh[4], *bh[4], *wm[4], *bm[4], *wo[4], *bo[4], *wl[4], *bl[4];
  for (int l = 0; l < 4; ++l) {
    wx[l] = (const float*)d_in[idx++]; bx[l] = (const float*)d_in[idx++];
    wh[l] = (const float*)d_in[idx++]; bh[l] = (const float*)d_in[idx++];
    wm[l] = (const float*)d_in[idx++]; bm[l] = (const float*)d_in[idx++];
    wo[l] = (const float*)d_in[idx++]; bo[l] = (const float*)d_in[idx++];
    wl[l] = (const float*)d_in[idx++]; bl[l] = (const float*)d_in[idx++];
  }
  // d_in[idx] = out_len (== 1), unused.

  // -------- workspace bump allocator --------
  char* wsp = (char*)d_ws;
  auto alloc = [&](size_t bytes) -> void* {
    void* p = (void*)wsp;
    wsp += (bytes + 255) & ~(size_t)255;
    return p;
  };

  // packed bf16 weights + BN scale/shift
  const int encCin[4]  = {3, 64, 64, 128};
  const int encCout[4] = {64, 64, 128, 128};
  unsigned short* pew[4]; int encKP[4];
  float *esc[4], *esh[4];
  for (int l = 0; l < 4; ++l) {
    int Ktot = encCin[l] * 9;
    encKP[l] = ((Ktot + 63) / 64) * 64;
    pew[l] = (unsigned short*)alloc((size_t)encCout[l] * encKP[l] * 2);
    esc[l] = (float*)alloc((size_t)encCout[l] * 4);
    esh[l] = (float*)alloc((size_t)encCout[l] * 4);
  }
  const int Kx = C * 9, Ko = 2 * C * 9, Kl = 2 * C;   // 1152, 2304, 256 (x64 multiples)
  unsigned short *pwx[4], *pwh[4], *pwm[4], *pwo[4], *pwl[4];
  for (int l = 0; l < 4; ++l) {
    pwx[l] = (unsigned short*)alloc((size_t)7 * C * Kx * 2);
    pwh[l] = (unsigned short*)alloc((size_t)4 * C * Kx * 2);
    pwm[l] = (unsigned short*)alloc((size_t)3 * C * Kx * 2);
    pwo[l] = (unsigned short*)alloc((size_t)C * Ko * 2);
    pwl[l] = (unsigned short*)alloc((size_t)C * Kl * 2);
  }

  // activations / state (fp32)
  const size_t N1 = (size_t)Bn * 64 * 128 * 128;   // encoder mid
  const size_t N  = (size_t)Bn * C * S * S;        // 8.39M
  float* e1   = (float*)alloc(N1 * 4);
  float* e2   = (float*)alloc(N1 * 4);
  float* inp  = (float*)alloc(N * 4);
  float* hs   = (float*)alloc(4 * N * 4);
  float* cs   = (float*)alloc(4 * N * 4);
  float* m    = (float*)alloc(N * 4);
  float* mem  = (float*)alloc(2 * N * 4);
  float* xcb  = (float*)alloc(7 * N * 4);
  float* hcb  = (float*)alloc(4 * N * 4);
  float* mcb  = (float*)alloc(3 * N * 4);
  float* opre = (float*)alloc(N * 4);
  float* ocb  = (float*)alloc(N * 4);
  float* hlb  = (float*)alloc(N * 4);

  // -------- weight packing + BN folding --------
  auto packg = [&](const float* w, unsigned short* wp, int Ktot, int KP, int Cout) {
    int total = Cout * KP;
    pack_w<<<(total + 255) / 256, 256, 0, stream>>>(w, wp, Ktot, KP, Cout);
  };
  for (int l = 0; l < 4; ++l) {
    packg(ew[l], pew[l], encCin[l] * 9, encKP[l], encCout[l]);
    bn_prep<<<(encCout[l] + 255) / 256, 256, 0, stream>>>(eb[l], eg[l], ebe[l], emu[l],
                                                          evr[l], esc[l], esh[l], encCout[l]);
  }
  for (int l = 0; l < 4; ++l) {
    packg(wx[l], pwx[l], Kx, Kx, 7 * C);
    packg(wh[l], pwh[l], Kx, Kx, 4 * C);
    packg(wm[l], pwm[l], Kx, Kx, 3 * C);
    packg(wo[l], pwo[l], Ko, Ko, C);
    packg(wl[l], pwl[l], Kl, Kl, C);
  }

  // -------- zero initial state --------
  fillz<<<(int)((4 * N + 255) / 256), 256, 0, stream>>>(hs, 4 * N);
  fillz<<<(int)((4 * N + 255) / 256), 256, 0, stream>>>(cs, 4 * N);
  fillz<<<(int)((N + 255) / 256), 256, 0, stream>>>(m, N);

  const int SS = S * S;
  const int gBlocks = (int)((N + 255) / 256);

  // -------- 4 timesteps --------
  for (int t = 0; t < 4; ++t) {
    // encoder: frame -> inp [B,128,64,64]
    launch_conv(stream, x, 12 * Himg * Himg, t * 3 * Himg * Himg,
                pew[0], esc[0], esh[0], e1, Bn, 3, 256, 256, 64, 3, 3, 2, 1, 1);
    launch_conv(stream, e1, 64 * 128 * 128, 0,
                pew[1], esc[1], esh[1], e2, Bn, 64, 128, 128, 64, 3, 3, 1, 1, 1);
    launch_conv(stream, e2, 64 * 128 * 128, 0,
                pew[2], esc[2], esh[2], e1, Bn, 64, 128, 128, 128, 3, 3, 2, 1, 1);
    launch_conv(stream, e1, 128 * 64 * 64, 0,
                pew[3], esc[3], esh[3], inp, Bn, 128, 64, 64, 128, 3, 3, 1, 1, 1);

    for (int li = 0; li < 4; ++li) {
      const float* xin = (li == 0) ? inp : (hs + (size_t)(li - 1) * N);
      float* hL = hs + (size_t)li * N;
      float* cL = cs + (size_t)li * N;

      launch_conv(stream, xin, C * SS, 0, pwx[li], bx[li], nullptr, xcb,
                  Bn, C, S, S, 7 * C, 3, 3, 1, 1, 0);
      launch_conv(stream, hL, C * SS, 0, pwh[li], bh[li], nullptr, hcb,
                  Bn, C, S, S, 4 * C, 3, 3, 1, 1, 0);
      launch_conv(stream, m, C * SS, 0, pwm[li], bm[li], nullptr, mcb,
                  Bn, C, S, S, 3 * C, 3, 3, 1, 1, 0);

      gates_kernel<<<gBlocks, 256, 0, stream>>>(xcb, hcb, mcb, cL, m, mem, opre, Bn, C, SS);

      launch_conv(stream, mem, 2 * C * SS, 0, pwo[li], bo[li], nullptr, ocb,
                  Bn, 2 * C, S, S, C, 3, 3, 1, 1, 0);
      launch_conv(stream, mem, 2 * C * SS, 0, pwl[li], bl[li], nullptr, hlb,
                  Bn, 2 * C, S, S, C, 1, 1, 1, 0, 0);

      hout_kernel<<<gBlocks, 256, 0, stream>>>(opre, ocb, hlb, hL, N);
    }
  }

  // -------- outputs: (hs[3], cs[3], mem) concatenated flat --------
  float* out = (float*)d_out;
  (void)hipMemcpyAsync(out,         hs + 3 * N, N * 4, hipMemcpyDeviceToDevice, stream);
  (void)hipMemcpyAsync(out + N,     cs + 3 * N, N * 4, hipMemcpyDeviceToDevice, stream);
  (void)hipMemcpyAsync(out + 2 * N, m,          N * 4, hipMemcpyDeviceToDevice, stream);
}
